// Synthesis_63797444215163
// MI455X (gfx1250) — matrix operand order repaired and checked by reading, not compiled, not run on hardware
//
#include <hip/hip_runtime.h>
#include <hip/hip_bf16.h>

// ---------------------------------------------------------------------------
// StyleGAN2 synthesis (B=8, 256x256) for MI455X / gfx1250.
//
// Feature chain runs f16 CHANNEL-LAST (B,H,W,C): every im2col B-column fill
// is one contiguous 16B load (or a gfx1250 async global->LDS b128 copy), and
// every WMMA output store is one packed 16B f16 store. Weights are modulated
// per layer into f16 TAP-MAJOR (K = t*I + i) so each 32-wide K-chunk has one
// uniform tap (no div/mod in the hot loop). Each wave computes a 16(outCh) x
// 32(pixel) tile = 2 x v_wmma_f32_16x16x32_f16 per A-fragment; 4 waves/block
// share a double-buffered LDS B-tile with one barrier per chunk.
// toRGB (O=3) uses an fp32-NCHW accumulate store mode; the RGB image chain
// (binomial up-FIR + accumulate) stays fp32 and ends directly in d_out.
// ---------------------------------------------------------------------------

typedef __attribute__((ext_vector_type(16))) _Float16 v16h;
typedef __attribute__((ext_vector_type(8)))  _Float16 v8h;
typedef __attribute__((ext_vector_type(8)))  float    v8f;
typedef int v4i __attribute__((vector_size(16)));

#define SQRT2F      1.41421356237309515f
#define INV_SQRT512 0.04419417382415922f
#define BT_STRIDE   40   // halves per B-tile column (32 + pad, keeps 16B align)

// gfx1250 async global->LDS copy path (ASYNCcnt), guarded so either toolchain
// (and the host pass) still compiles via the plain load/ds_store fallback.
// Probe-learned signature: (v4i AS(1)*, v4i AS(3)*, imm offset, imm cpol).
#if defined(__AMDGCN__) && __has_builtin(__builtin_amdgcn_global_load_async_to_lds_b128)
#define HAVE_ASYNC_LDS 1
typedef __attribute__((address_space(1))) v4i* gptr_v4i;
typedef __attribute__((address_space(3))) v4i* lptr_v4i;
#else
#define HAVE_ASYNC_LDS 0
#endif

__device__ __forceinline__ void async_fence() {
#if HAVE_ASYNC_LDS
#if __has_builtin(__builtin_amdgcn_s_wait_asynccnt)
  __builtin_amdgcn_s_wait_asynccnt(0);
#else
  asm volatile("s_wait_asynccnt 0x0" ::: "memory");
#endif
#endif
}

// ---------------------------------------------------------------------------
// Broadcast the learned constant (512,4,4 fp32 CHW) to (8,4,4,512) f16 CL.
__global__ void bcast_cl_kernel(const float* __restrict__ src,
                                _Float16* __restrict__ dst, int total) {
  int idx = blockIdx.x * blockDim.x + threadIdx.x;   // ((b*16)+pix)*512 + c
  if (idx >= total) return;
  int c   = idx & 511;
  int pix = (idx >> 9) & 15;
  dst[idx] = (_Float16)src[c * 16 + pix];
}

// ---------------------------------------------------------------------------
// s[b,i] = (style[b,:] . aff_w[i,:]) / sqrt(512) + aff_b[i] + 1
__global__ void affine_kernel(const float* __restrict__ wsty,   // (B,512)
                              const float* __restrict__ aw,     // (I,512)
                              const float* __restrict__ ab,     // (I)
                              float* __restrict__ s, int I) {
  int i = blockIdx.x * blockDim.x + threadIdx.x;
  int b = blockIdx.y;
  if (i >= I) return;
  const float* wrow = wsty + (size_t)b * 512;
  const float* arow = aw + (size_t)i * 512;
  float acc = 0.f;
  for (int d = 0; d < 512; ++d) acc += wrow[d] * arow[d];
  s[b * I + i] = acc * INV_SQRT512 + ab[i] + 1.0f;
}

// ---------------------------------------------------------------------------
// Per-(b,o): w = cw * s * 1/sqrt(I*taps); optional demod by rsqrt(sum w^2).
// Output f16, TAP-MAJOR: wmod[b][o][t*I + i]; optional spatial flip
// (conv_transpose pre-flip: t -> taps-1-t).
__global__ __launch_bounds__(256)
void modulate_kernel(const float* __restrict__ cw,   // (O, I, taps)
                     const float* __restrict__ s,    // (B, I)
                     _Float16* __restrict__ wmod,    // (B, O, taps*I)
                     int I, int O, int taps, int demod, int flip) {
  int o = blockIdx.x, b = blockIdx.y;
  int K = I * taps;
  float scale = rsqrtf((float)K);
  const float* crow = cw + (size_t)o * K;
  const float* srow = s + (size_t)b * I;
  float ss = 0.f;
  for (int e = threadIdx.x; e < K; e += 256) {
    int i = e / taps;
    float v = crow[e] * srow[i] * scale;
    ss += v * v;
  }
  __shared__ float red[256];
  red[threadIdx.x] = ss;
  __syncthreads();
  for (int st = 128; st > 0; st >>= 1) {
    if (threadIdx.x < st) red[threadIdx.x] += red[threadIdx.x + st];
    __syncthreads();
  }
  float d = demod ? rsqrtf(red[0] + 1e-8f) : 1.0f;
  _Float16* wrow = wmod + ((size_t)b * O + o) * K;
  for (int e = threadIdx.x; e < K; e += 256) {
    int i = e / taps, t = e - i * taps;
    float v = crow[e] * srow[i] * scale * d;
    int tt = flip ? (taps - 1 - t) : t;
    wrow[tt * I + i] = (_Float16)v;     // tap-major
  }
}

// ---------------------------------------------------------------------------
// Fill one 32K x 32N B-tile chunk (tap t, channels i0..i0+31) into LDS.
// Channel-last f16 input: each of the 128 threads owns one column (n=tid&31)
// and 8 contiguous channels -> one async b128 global->LDS copy (or one 16B
// load + ds_store_b128). Tap/pixel validity resolved once per chunk.
__device__ __forceinline__
void fill_tile(_Float16* __restrict__ dst, const _Float16* __restrict__ xin,
               int b, int I, int H, int W, int OW, int Ntot,
               int taps, int dilated, int pb, int i0, int t, int tid) {
  const int n  = tid & 31;
  const int kc = (tid >> 5) * 8;
  const int p  = pb + n;
  const _Float16* src = nullptr;
  if (p < Ntot) {
    int py = p / OW, px = p - py * OW;
    int ky = t / 3, kx = t - ky * 3;
    if (taps == 1) { ky = 0; kx = 0; }
    if (!dilated) {
      int pad = (taps == 9) ? 1 : 0;
      int iy = py + ky - pad, ix = px + kx - pad;
      if (iy >= 0 && iy < H && ix >= 0 && ix < W)
        src = xin + ((size_t)(b * H + iy) * W + ix) * I;
    } else {
      int qy = py + ky - 2, qx = px + kx - 2;     // pad = k-1 = 2
      if (qy >= 0 && qx >= 0 && !(qy & 1) && !(qx & 1)) {
        qy >>= 1; qx >>= 1;                       // zero-stuffed x2 input
        if (qy < H && qx < W)
          src = xin + ((size_t)(b * H + qy) * W + qx) * I;
      }
    }
  }
  _Float16* ld = &dst[n * BT_STRIDE + kc];
  if (src) {
#if HAVE_ASYNC_LDS
    __builtin_amdgcn_global_load_async_to_lds_b128(
        (gptr_v4i)(src + i0 + kc), (lptr_v4i)ld, 0, 0);
#else
    *(v8h*)ld = *(const v8h*)(src + i0 + kc);     // 16B load + ds_store_b128
#endif
  } else {
    *(v8h*)ld = (v8h){};
  }
}

// ---------------------------------------------------------------------------
// Implicit-GEMM modulated conv on WMMA. xin is f16 channel-last (B,H,W,I).
//   taps==9, dilated==0 : 3x3 conv, pad 1, out H x W
//   taps==9, dilated==1 : conv_transpose(stride2) as lhs-dilated conv with
//                         pre-flipped weights, out (2H+1) x (2W+1)
//   taps==1             : 1x1 conv (toRGB), out H x W
// mode 0: out f16 channel-last (B,OH,OW,O), packed v8h stores
// mode 1: out fp32 NCHW, accumulate (toRGB into upsampled image)
// mode 2: out fp32 NCHW, overwrite (first toRGB)
__global__ __launch_bounds__(128)
void modconv_wmma(const _Float16* __restrict__ xin,
                  const _Float16* __restrict__ wmod,   // (B, O, taps*I)
                  void* __restrict__ yout,
                  int I, int O, int H, int W,
                  int taps, int dilated, int mode) {
  const int b    = blockIdx.z;
  const int OH   = dilated ? 2 * H + 1 : H;
  const int OW   = dilated ? 2 * W + 1 : W;
  const int Ntot = OH * OW;
  const int Ktot = I * taps;
  const int tid  = threadIdx.x;
  const int lane = tid & 31;
  const int wave = tid >> 5;
  const int hi   = lane >> 4;    // half-wave select for K striping
  const int ln   = lane & 15;    // row (A) / column (B,D) index
  const int ocb  = (blockIdx.y * 4 + wave) * 16;
  const int pb   = blockIdx.x * 32;

  __shared__ _Float16 Bt[2][32 * BT_STRIDE];

  const int  oc     = ocb + ln;
  const bool wvalid = (oc < O);
  const _Float16* wrow = wmod + ((size_t)b * O + (wvalid ? oc : 0)) * Ktot;

  // Chunk c covers K = [32c, 32c+32): tap t = c >> lg, channels i0 = (c&m)<<5.
  const int nc = I >> 5;                 // chunks per tap (power of two)
  const int m  = nc - 1;
  const int lg = __ffs(nc) - 1;
  const int C  = taps * nc;

  fill_tile(Bt[0], xin, b, I, H, W, OW, Ntot, taps, dilated, pb, 0, 0, tid);

  v8f acc0 = {}, acc1 = {};
  int buf = 0;
  for (int c = 0; c < C; ++c) {
    async_fence();                       // our async fills have landed in LDS
    __syncthreads();                     // tile c visible; tile c-1 consumed
    if (c + 1 < C) {
      int c1 = c + 1;
      fill_tile(Bt[buf ^ 1], xin, b, I, H, W, OW, Ntot, taps, dilated, pb,
                (c1 & m) << 5, c1 >> lg, tid);
    }
    // A fragment (tap-major weights => K base is just 32c):
    //   halves j -> K = 32c + hi*16 + j (contiguous, matches B).
    const int kbase = c << 5;
    v8h alo = {}, ahi = {};
    if (wvalid) {
      alo = *(const v8h*)(wrow + kbase + hi * 16);
      ahi = *(const v8h*)(wrow + kbase + hi * 16 + 8);
      __builtin_prefetch(wrow + kbase + 32, 0, 1);   // global_prefetch_b8
    }
    v16h a = {};
#pragma unroll
    for (int j = 0; j < 8; ++j) { a[j] = alo[j]; a[j + 8] = ahi[j]; }

    // Two B fragments from LDS (columns ln and ln+16), K = hi*16 + j,
    // contiguous per lane -> ds_load_b128 pairs.
    const _Float16* bt = Bt[buf];
    v8h b0l = *(const v8h*)&bt[(ln)      * BT_STRIDE + hi * 16];
    v8h b0h = *(const v8h*)&bt[(ln)      * BT_STRIDE + hi * 16 + 8];
    v8h b1l = *(const v8h*)&bt[(ln + 16) * BT_STRIDE + hi * 16];
    v8h b1h = *(const v8h*)&bt[(ln + 16) * BT_STRIDE + hi * 16 + 8];
    v16h b0 = {}, b1 = {};
#pragma unroll
    for (int j = 0; j < 8; ++j) {
      b0[j] = b0l[j]; b0[j + 8] = b0h[j];
      b1[j] = b1l[j]; b1[j + 8] = b1h[j];
    }

    acc0 = __builtin_amdgcn_wmma_f32_16x16x32_f16(false, a, false, b0,
                                                  (short)0, acc0, false, false);
    acc1 = __builtin_amdgcn_wmma_f32_16x16x32_f16(false, a, false, b1,
                                                  (short)0, acc1, false, false);
    buf ^= 1;
  }

  // D layout: column N = ln (+16 for acc1), row M = v + 8*hi per VGPR v,
  // i.e. channels [ocb + 8*hi, +8) -- contiguous in channel-last memory.
  const int p0 = pb + ln, p1 = pb + 16 + ln;
  if (mode == 0) {
    _Float16* yh = (_Float16*)yout;
    if (ocb + hi * 8 < O) {
      v8h o0, o1;
#pragma unroll
      for (int v = 0; v < 8; ++v) { o0[v] = (_Float16)acc0[v]; o1[v] = (_Float16)acc1[v]; }
      size_t cb = (size_t)(ocb + hi * 8);
      if (p0 < Ntot) *(v8h*)&yh[((size_t)b * Ntot + p0) * O + cb] = o0;
      if (p1 < Ntot) *(v8h*)&yh[((size_t)b * Ntot + p1) * O + cb] = o1;
    }
  } else {
    float* yf = (float*)yout;
#pragma unroll
    for (int v = 0; v < 8; ++v) {
      int occ = ocb + v + hi * 8;
      if (occ < O) {
        size_t rowb = ((size_t)b * O + occ) * (size_t)Ntot;
        if (p0 < Ntot) {
          float r = acc0[v];
          if (mode == 1) r += yf[rowb + p0];
          yf[rowb + p0] = r;
        }
        if (p1 < Ntot) {
          float r = acc1[v];
          if (mode == 1) r += yf[rowb + p1];
          yf[rowb + p1] = r;
        }
      }
    }
  }
}

// ---------------------------------------------------------------------------
// Channel-last f16 binomial FIR, up=1, pad=1, gain=4:
// post-conv_transpose smoothing, (B,2H+1,2W+1,C) -> (B,2H,2W,C).
__global__ void fir_cl_kernel(const _Float16* __restrict__ xin,
                              _Float16* __restrict__ yout,
                              int C, int inH, int inW, int outH, int outW,
                              long total) {
  long idx = blockIdx.x * (long)blockDim.x + threadIdx.x;
  if (idx >= total) return;
  int  c  = (int)(idx % C);
  long r1 = idx / C;
  int  ox = (int)(r1 % outW);
  long r2 = r1 / outW;
  int  oy = (int)(r2 % outH);
  int  b  = (int)(r2 / outH);
  const float cf[4] = {0.125f, 0.375f, 0.375f, 0.125f};
  float acc = 0.f;
#pragma unroll
  for (int a = 0; a < 4; ++a) {
    int qy = oy + a - 1;
    if (qy < 0 || qy >= inH) continue;
#pragma unroll
    for (int bb = 0; bb < 4; ++bb) {
      int qx = ox + bb - 1;
      if (qx < 0 || qx >= inW) continue;
      acc += cf[a] * cf[bb] *
             (float)xin[((size_t)(b * inH + qy) * inW + qx) * C + c];
    }
  }
  yout[idx] = (_Float16)(acc * 4.0f);
}

// ---------------------------------------------------------------------------
// fp32 NCHW binomial FIR with x2 zero-stuffing (up=2,pad=2,gain=4):
// upsample2d of the RGB image, (B*3,H,W) -> (B*3,2H,2W).
__global__ void fir_up_kernel(const float* __restrict__ xin,
                              float* __restrict__ yout,
                              int inH, int inW, int outH, int outW, long total) {
  long idx = blockIdx.x * (long)blockDim.x + threadIdx.x;
  if (idx >= total) return;
  int  ox = (int)(idx % outW);
  long r1 = idx / outW;
  int  oy = (int)(r1 % outH);
  long pl = r1 / outH;                       // plane = b*3 + c
  const float cf[4] = {0.125f, 0.375f, 0.375f, 0.125f};
  const float* plane = xin + pl * (long)inH * inW;
  float acc = 0.f;
#pragma unroll
  for (int a = 0; a < 4; ++a) {
    int q = oy + a - 2;
    if (q < 0 || (q & 1)) continue;
    int qy = q >> 1;
    if (qy >= inH) continue;
#pragma unroll
    for (int bb = 0; bb < 4; ++bb) {
      int r = ox + bb - 2;
      if (r < 0 || (r & 1)) continue;
      int qx = r >> 1;
      if (qx >= inW) continue;
      acc += cf[a] * cf[bb] * plane[qy * (long)inW + qx];
    }
  }
  yout[idx] = acc * 4.0f;
}

// ---------------------------------------------------------------------------
// Channel-last f16: y = leaky_relu(y + noise*nscale + bias, 0.2)*sqrt(2).
__global__ void nbl_cl_kernel(_Float16* __restrict__ y,
                              const float* __restrict__ noise,
                              const float* __restrict__ nscale,
                              const float* __restrict__ bias,
                              int O, int HW, long total) {
  long idx = blockIdx.x * (long)blockDim.x + threadIdx.x;
  if (idx >= total) return;
  int  c = (int)(idx % O);
  long r = idx / O;
  int  p = (int)(r % HW);
  long b = r / HW;
  float v = (float)y[idx] + noise[b * (long)HW + p] * nscale[0] + bias[c];
  y[idx] = (_Float16)((v < 0.f ? 0.2f * v : v) * SQRT2F);
}

// ---------------------------------------------------------------------------
extern "C" void kernel_launch(void* const* d_in, const int* in_sizes, int n_in,
                              void* d_out, int out_size, void* d_ws, size_t ws_size,
                              hipStream_t stream) {
  (void)in_sizes; (void)n_in; (void)out_size; (void)ws_size;

  // Input flattening (pytree order, dict keys sorted):
  //   0: w_all (13,8,512); 1..13: noise0..12; 14: const (1,512,4,4);
  //   15+5l+{0..4}: layer l {aff_b, aff_w, bias, cw, nscale};
  //   80+3t+{0..2}: torgb t {aff_b, aff_w, cw}.
  const float* w_all = (const float*)d_in[0];
  const float* noise[13];
  for (int j = 0; j < 13; ++j) noise[j] = (const float*)d_in[1 + j];
  const float* cconst = (const float*)d_in[14];
  auto LP = [&](int l, int w) { return (const float*)d_in[15 + 5 * l + w]; };
  auto TP = [&](int t, int w) { return (const float*)d_in[80 + 3 * t + w]; };

  // Workspace carve-up (16B-aligned for the f16 vector paths).
  char* ws = (char*)d_ws;
  float*    s_buf = (float*)ws;                           // 16 KB
  _Float16* wmod  = (_Float16*)(ws + (16u << 10));        // 8*512*4608 f16
  _Float16* actA  = (_Float16*)(ws + 16384 + 37748736);   // f16 CL activations
  _Float16* actB  = actA + 16777216;                      // f16 CL activations
  _Float16* zbuf  = actB + 16777216;                      // dilated conv output
  float*    img0  = (float*)(zbuf + 16909312);            // RGB ping (fp32 NCHW)
  float*    img1  = img0 + 1572864;                       // RGB pong

  auto launch_conv = [&](const _Float16* xin, void* yout, int I, int O,
                         int H, int W, int taps, int dilated, int mode) {
    int OH = dilated ? 2 * H + 1 : H;
    int OW = dilated ? 2 * W + 1 : W;
    int Ntot = OH * OW;
    dim3 grid((unsigned)((Ntot + 31) / 32), (unsigned)((O + 63) / 64), 8);
    modconv_wmma<<<grid, 128, 0, stream>>>(xin, wmod, yout, I, O, H, W,
                                           taps, dilated, mode);
  };

  auto launch_style = [&](int l, const _Float16* xin, _Float16* yout,
                          int I, int O, int res, int up) {
    dim3 ag((unsigned)((I + 255) / 256), 8);
    affine_kernel<<<ag, 256, 0, stream>>>(w_all + (size_t)l * 8 * 512,
                                          LP(l, 1), LP(l, 0), s_buf, I);
    dim3 mg((unsigned)O, 8);
    modulate_kernel<<<mg, 256, 0, stream>>>(LP(l, 3), s_buf, wmod, I, O, 9, 1,
                                            up == 2 ? 1 : 0);
    if (up == 1) {
      launch_conv(xin, yout, I, O, res, res, 9, 0, 0);
      long total = 8L * O * res * res;
      nbl_cl_kernel<<<(unsigned)((total + 255) / 256), 256, 0, stream>>>(
          yout, noise[l], LP(l, 4), LP(l, 2), O, res * res, total);
    } else {
      launch_conv(xin, zbuf, I, O, res, res, 9, 1, 0);      // (2r+1)^2, f16 CL
      int nr = 2 * res;
      long total = 8L * O * nr * nr;
      fir_cl_kernel<<<(unsigned)((total + 255) / 256), 256, 0, stream>>>(
          zbuf, yout, O, 2 * res + 1, 2 * res + 1, nr, nr, total);
      nbl_cl_kernel<<<(unsigned)((total + 255) / 256), 256, 0, stream>>>(
          yout, noise[l], LP(l, 4), LP(l, 2), O, nr * nr, total);
    }
  };

  auto launch_torgb = [&](int t, int styleLayer, const _Float16* xin, float* img,
                          int I, int res, int accum) {
    dim3 ag((unsigned)((I + 255) / 256), 8);
    affine_kernel<<<ag, 256, 0, stream>>>(w_all + (size_t)styleLayer * 8 * 512,
                                          TP(t, 1), TP(t, 0), s_buf, I);
    dim3 mg(3, 8);
    modulate_kernel<<<mg, 256, 0, stream>>>(TP(t, 2), s_buf, wmod, I, 3, 1, 0, 0);
    launch_conv(xin, img, I, 3, res, res, 1, 0, accum ? 1 : 2);
  };

  // x = broadcast(const) over batch, transposed to channel-last f16.
  bcast_cl_kernel<<<(8 * 16 * 512 + 255) / 256, 256, 0, stream>>>(
      cconst, actA, 8 * 16 * 512);
  _Float16* cur = actA;
  _Float16* nxt = actB;

  // Layer 0 + first toRGB at res 4.
  launch_style(0, cur, nxt, 512, 512, 4, 1);
  { _Float16* t = cur; cur = nxt; nxt = t; }
  launch_torgb(0, 0, cur, img0, 512, 4, 0);
  float* img = img0;

  static const int IC[6] = {512, 512, 512, 256, 128, 64};
  static const int OC[6] = {512, 512, 256, 128, 64, 32};
  int res = 4;
  for (int blk = 0; blk < 6; ++blk) {
    int l1 = 1 + 2 * blk, l2 = 2 + 2 * blk;
    launch_style(l1, cur, nxt, IC[blk], OC[blk], res, 1);
    { _Float16* t = cur; cur = nxt; nxt = t; }
    launch_style(l2, cur, nxt, OC[blk], OC[blk], res, 2);
    { _Float16* t = cur; cur = nxt; nxt = t; }
    int nres = res * 2;
    // Final block writes straight into d_out (fir fully overwrites it first).
    float* dst = (blk == 5) ? (float*)d_out : ((img == img0) ? img1 : img0);
    long total = 8L * 3 * nres * nres;
    fir_up_kernel<<<(unsigned)((total + 255) / 256), 256, 0, stream>>>(
        img, dst, res, res, nres, nres, total);
    launch_torgb(blk + 1, l2, cur, dst, OC[blk], nres, 1);
    img = dst;
    res = nres;
  }
}
